// GenView_1778116461037
// MI455X (gfx1250) — matrix operand order repaired and checked by reading, not compiled, not run on hardware
//
#include <hip/hip_runtime.h>
#include <math.h>

// ---------------------------------------------------------------------------
// GenView: 1-layer GCN encoder + per-edge MLP + segment softmax (CDNA5/gfx1250)
//   xw   = feat @ W_gcn + b_gcn            (WMMA f16->f32, 16x16x32 tiles)
//   emb  = segment_sum(v * xw[cols], rows) (float4 global atomics, L2-resident)
//   temp = [emb[r]|emb[c]] @ W_mlp + b     (per-edge 128-wide dot)
//   pi   = segment_softmax(temp, rows)
//   out  = v_ori + 0.5 * pi
// ---------------------------------------------------------------------------

#define F_IN 256
#define HID  64
#define COM_LAMBDA 0.5f

typedef __attribute__((ext_vector_type(16))) _Float16 v16h;
typedef __attribute__((ext_vector_type(8)))  float    v8f;

// ---------------------------------------------------------------------------
__global__ void init_ws(float* __restrict__ emb, float* __restrict__ segmax,
                        float* __restrict__ segsum, int n, int nh) {
    int i = blockIdx.x * blockDim.x + threadIdx.x;
    if (i < nh) emb[i] = 0.0f;
    if (i < n) { segmax[i] = -INFINITY; segsum[i] = 0.0f; }
}

// ---------------------------------------------------------------------------
// One wave32 computes one 16x16 tile of xw = feat @ W_gcn + b_gcn.
// K loop: F_IN/32 = 8 iterations of v_wmma_f32_16x16x32_f16.
// Fragment layouts per CDNA5 ISA 7.12.2 (16-bit A 16x32; B mirrored; f32 C/D).
__global__ void gemm_wmma(const float* __restrict__ feat,
                          const float* __restrict__ Wg,
                          const float* __restrict__ bg,
                          float* __restrict__ xw, int tiles_m) {
    const int wave = (blockIdx.x * blockDim.x + threadIdx.x) >> 5;
    const int lane = threadIdx.x & 31;
    const int tile_total = tiles_m * (HID / 16);
    if (wave >= tile_total) return;

    const int tm = wave >> 2;          // HID/16 == 4 tiles along N
    const int tn = wave & 3;
    const int half = lane >> 4;        // 0: lanes 0-15, 1: lanes 16-31
    const int lp   = lane & 15;
    const int m0 = tm * 16, n0 = tn * 16;

    v8f c = {};
    for (int k0 = 0; k0 < F_IN; k0 += 32) {
        v16h a, b;
#pragma unroll
        for (int v = 0; v < 8; ++v) {
            // A (16x32, MxK): VGPR0-3 -> K 0..7 / 8..15 ; VGPR4-7 -> K 16..23 / 24..31
            const int ka = k0 + 2 * v + 8 * half + ((v >= 4) ? 8 : 0);
            // B (32x16, KxN): same K order as A, col n=lp
            const int kb = k0 + 2 * v + 8 * half + ((v >= 4) ? 8 : 0);
            a[2 * v]     = (_Float16)feat[(size_t)(m0 + lp) * F_IN + ka];
            a[2 * v + 1] = (_Float16)feat[(size_t)(m0 + lp) * F_IN + ka + 1];
            b[2 * v]     = (_Float16)Wg[(size_t)kb * HID + (n0 + lp)];
            b[2 * v + 1] = (_Float16)Wg[(size_t)(kb + 1) * HID + (n0 + lp)];
        }
        c = __builtin_amdgcn_wmma_f32_16x16x32_f16(
                /*neg_a=*/false, a, /*neg_b=*/false, b,
                /*c_mod=*/(short)0, c, /*reuse_a=*/false, /*reuse_b=*/false);
    }

    const float bias = bg[n0 + lp];
#pragma unroll
    for (int v = 0; v < 8; ++v)   // C/D: VGPR v -> row M = v (+8 for upper half)
        xw[(size_t)(m0 + v + 8 * half) * HID + (n0 + lp)] = c[v] + bias;
}

// ---------------------------------------------------------------------------
// emb[rows[e]] += v[e] * xw[cols[e]]  -- 16 threads per edge, float4 lanes.
__global__ void scatter_msg(const float* __restrict__ xw,
                            const float* __restrict__ vv,
                            const int* __restrict__ rows,
                            const int* __restrict__ cols,
                            float* __restrict__ emb, int E) {
    const long long gid = (long long)blockIdx.x * blockDim.x + threadIdx.x;
    const long long e = gid >> 4;
    if (e >= E) return;
    const int j = (int)(gid & 15) * 4;
    const int r = rows[e], c = cols[e];
    const float s = vv[e];
    const float4 x = *(const float4*)(xw + (size_t)c * HID + j);
    float* dst = emb + (size_t)r * HID + j;
    atomicAdd(dst + 0, s * x.x);
    atomicAdd(dst + 1, s * x.y);
    atomicAdd(dst + 2, s * x.z);
    atomicAdd(dst + 3, s * x.w);
}

// ---------------------------------------------------------------------------
__device__ __forceinline__ void atomicMaxF32(float* addr, float val) {
    // IEEE ordering trick: signed-int max for non-negative, unsigned min for negative.
    if (val >= 0.0f) atomicMax((int*)addr, __float_as_int(val));
    else             atomicMin((unsigned int*)addr, __float_as_uint(val));
}

// temp[e] = dot(emb[r], Wm[0:64]) + dot(emb[c], Wm[64:128]) + bm ; segmax via atomics.
__global__ void edge_mlp(const float* __restrict__ emb,
                         const float* __restrict__ Wm,
                         const float* __restrict__ bm,
                         const int* __restrict__ rows,
                         const int* __restrict__ cols,
                         float* __restrict__ temp,
                         float* __restrict__ segmax, int E) {
    const long long e = (long long)blockIdx.x * blockDim.x + threadIdx.x;
    if (e >= E) return;
    const int r = rows[e], c = cols[e];
    const float4* er = (const float4*)(emb + (size_t)r * HID);
    const float4* ec = (const float4*)(emb + (size_t)c * HID);
    const float4* w0 = (const float4*)(Wm);
    const float4* w1 = (const float4*)(Wm + HID);
    float acc = bm[0];
#pragma unroll
    for (int i = 0; i < HID / 4; ++i) {
        const float4 a = er[i], wa = w0[i];
        const float4 b = ec[i], wb = w1[i];
        acc += a.x * wa.x + a.y * wa.y + a.z * wa.z + a.w * wa.w;
        acc += b.x * wb.x + b.y * wb.y + b.z * wb.z + b.w * wb.w;
    }
    temp[e] = acc;
    atomicMaxF32(segmax + r, acc);
}

// ---------------------------------------------------------------------------
__global__ void edge_exp(const float* __restrict__ segmax,
                         const int* __restrict__ rows,
                         float* __restrict__ temp,
                         float* __restrict__ segsum, int E) {
    const long long e = (long long)blockIdx.x * blockDim.x + threadIdx.x;
    if (e >= E) return;
    const int r = rows[e];
    const float v = __expf(temp[e] - segmax[r]);
    temp[e] = v;
    atomicAdd(segsum + r, v);
}

// ---------------------------------------------------------------------------
__global__ void finalize(const float* __restrict__ vv,
                         const float* __restrict__ temp,
                         const float* __restrict__ segsum,
                         const int* __restrict__ rows,
                         float* __restrict__ out, int E) {
    const long long e = (long long)blockIdx.x * blockDim.x + threadIdx.x;
    if (e >= E) return;
    out[e] = vv[e] + COM_LAMBDA * (temp[e] / segsum[rows[e]]);
}

// ---------------------------------------------------------------------------
extern "C" void kernel_launch(void* const* d_in, const int* in_sizes, int n_in,
                              void* d_out, int out_size, void* d_ws, size_t ws_size,
                              hipStream_t stream) {
    const float* v_ori = (const float*)d_in[0];
    const float* feat  = (const float*)d_in[1];
    const float* W_gcn = (const float*)d_in[2];
    const float* b_gcn = (const float*)d_in[3];
    const float* W_mlp = (const float*)d_in[4];
    const float* b_mlp = (const float*)d_in[5];
    const int*   vidx  = (const int*)d_in[6];

    const int E = in_sizes[0];
    const int N = in_sizes[1] / F_IN;
    const int* rows = vidx;
    const int* cols = vidx + E;

    // Workspace layout (floats): xw[N*HID] | emb[N*HID] | temp[E] | segmax[N] | segsum[N]
    float* xw     = (float*)d_ws;
    float* emb    = xw + (size_t)N * HID;
    float* temp   = emb + (size_t)N * HID;
    float* segmax = temp + (size_t)E;
    float* segsum = segmax + (size_t)N;

    const int BLK = 256;

    // 1) init emb / segmax / segsum
    {
        int work = N * HID;
        init_ws<<<(work + BLK - 1) / BLK, BLK, 0, stream>>>(emb, segmax, segsum, N, work);
    }
    // 2) xw = feat @ W_gcn + b_gcn  (WMMA)
    {
        const int tiles_m = (N + 15) / 16;
        const long long threads = (long long)tiles_m * (HID / 16) * 32;
        gemm_wmma<<<(int)((threads + BLK - 1) / BLK), BLK, 0, stream>>>(feat, W_gcn, b_gcn, xw, tiles_m);
    }
    // 3) emb = segment_sum(v * xw[cols], rows)
    {
        const long long threads = (long long)E * 16;
        scatter_msg<<<(int)((threads + BLK - 1) / BLK), BLK, 0, stream>>>(xw, v_ori, rows, cols, emb, E);
    }
    // 4) per-edge MLP + segment max
    edge_mlp<<<(E + BLK - 1) / BLK, BLK, 0, stream>>>(emb, W_mlp, b_mlp, rows, cols, temp, segmax, E);
    // 5) exp + segment sum
    edge_exp<<<(E + BLK - 1) / BLK, BLK, 0, stream>>>(segmax, rows, temp, segsum, E);
    // 6) out = v_ori + lambda * pi
    finalize<<<(E + BLK - 1) / BLK, BLK, 0, stream>>>(v_ori, temp, segsum, rows, (float*)d_out, E);
}
